// MultiHeadAttention_10247791968652
// MI455X (gfx1250) — compile-verified, matrix operand order checked by reading, not run on hardware
//
#include <hip/hip_runtime.h>
#include <hip/hip_bf16.h>

// ---------------------------------------------------------------------------
// MHA block for MI455X (gfx1250, wave32):
//   - all GEMMs via v_wmma_f32_16x16x32_bf16 (f32 accumulate)
//   - tiles fed by TDM (tensor_load_to_lds, TENSORcnt) and
//     global_load_async_to_lds_b128 (ASYNCcnt) -- no VGPR round-trips
//   - every kernel double-buffers DMA against WMMA compute (unroll 1 so the
//     hand pipeline keeps accumulators in place: no v_mov/v_nop hazard chains)
//   - per-head LayerNorm + softmax row stats done in registers via shfl_xor
// ---------------------------------------------------------------------------

typedef __bf16 bf16;
typedef __attribute__((ext_vector_type(8)))  __bf16 v8bf;
typedef __attribute__((ext_vector_type(16))) __bf16 v16bf;
typedef __attribute__((ext_vector_type(8)))  float  v8f;
typedef __attribute__((ext_vector_type(4)))  unsigned int u32x4;
typedef __attribute__((ext_vector_type(8)))  unsigned int u32x8;

#define NB 16
#define NN 1024
#define ND 768
#define NH 12
#define HD 64
#define BN (NB * NN)          // 16384 tokens
#define SCALE 0.125f          // 64^-0.5
#define LN_EPS 1e-5f
#define KB 64                 // keys per attention block
#define NBLK (NN / KB)        // 16

// ---- LDS offset of a __shared__ object (LDS aperture: addr[31:0] = offset) -
__device__ __forceinline__ unsigned int lds_off(const void* p) {
  return (unsigned int)(unsigned long long)p;
}

// ---- Per-lane async copy global -> LDS (16B), tracked by ASYNCcnt ---------
__device__ __forceinline__ void async_ld_b128(const bf16* lds_dst, const bf16* gsrc) {
  unsigned int dst = lds_off(lds_dst);
  asm volatile("global_load_async_to_lds_b128 %0, %1, off"
               :: "v"(dst), "v"(gsrc) : "memory");
}
__device__ __forceinline__ void wait_async0() {
  asm volatile("s_wait_asynccnt 0x0" ::: "memory");
}
__device__ __forceinline__ void wait_tensor0() {
  asm volatile("s_wait_tensorcnt 0x0" ::: "memory");
}

// ---- TDM: 2-D tile load global -> LDS with LDS padding (TENSORcnt) --------
// pad_int code: 3 -> every 16 DWORDs (64B row), 4 -> every 32 DWORDs (128B row)
// pad_amt code: 3 -> 4 DWORDs (16B) => LDS row pitch = row + 16B
__device__ __forceinline__ void tdm_load_2d(unsigned int lds_addr, const void* gptr,
                                            unsigned int tile_d0, unsigned int tile_d1,
                                            unsigned int tensor_d0, unsigned int tensor_d1,
                                            unsigned int stride_elems,
                                            unsigned int pad_int, unsigned int pad_amt) {
  unsigned long long ga = (unsigned long long)gptr;
  u32x4 g0;
  g0[0] = 1u;                                      // count=1, user descriptor
  g0[1] = lds_addr;                                // lds_addr (bytes)
  g0[2] = (unsigned int)ga;                        // global_addr[31:0]
  g0[3] = (unsigned int)(ga >> 32) | 0x80000000u;  // global_addr[56:32] | type=2
  u32x8 g1;
  g1[0] = (1u << 16) | (1u << 20) | (pad_int << 22) | (pad_amt << 25); // 2B, pad
  g1[1] = (tensor_d0 & 0xFFFFu) << 16;
  g1[2] = (tensor_d0 >> 16) | ((tensor_d1 & 0xFFFFu) << 16);
  g1[3] = (tensor_d1 >> 16) | (tile_d0 << 16);
  g1[4] = tile_d1 & 0xFFFFu;                       // tile_dim1 (tile_dim2 = 0)
  g1[5] = stride_elems;                            // tensor_dim0_stride[31:0]
  g1[6] = 0;
  g1[7] = 0;
  asm volatile("tensor_load_to_lds %0, %1" :: "s"(g0), "s"(g1) : "memory");
}

// ---- WMMA fragment loader (ISA 7.12.2 16-bit A layout; B loaded from B^T) -
__device__ __forceinline__ v16bf load_frag(const bf16* base, int stride, int lane) {
  const int row  = lane & 15;
  const int half = lane >> 4;
  const bf16* p = base + row * stride + half * 8;
  v8bf lo = *reinterpret_cast<const v8bf*>(p);
  v8bf hi = *reinterpret_cast<const v8bf*>(p + 16);
  v16bf r;
#pragma unroll
  for (int i = 0; i < 8; ++i) { r[i] = lo[i]; r[i + 8] = hi[i]; }
  return r;
}

__device__ __forceinline__ v8f wmma_bf16(v16bf a, v16bf b, v8f c) {
  return __builtin_amdgcn_wmma_f32_16x16x32_bf16(false, a, false, b, (short)0, c,
                                                 false, false);
}

// ---- Kernel 1: LayerNorm(x) over D=768, output bf16 -----------------------
__global__ void k_ln_x(const float* __restrict__ x, const float* __restrict__ g,
                       const float* __restrict__ b, bf16* __restrict__ xn) {
  const int m = blockIdx.x;
  const int tid = threadIdx.x;          // 256 threads, 3 elems each
  __shared__ float s1[256], s2[256];
  const float* xr = x + (size_t)m * ND;
  float a0 = xr[tid], a1 = xr[tid + 256], a2 = xr[tid + 512];
  s1[tid] = a0 + a1 + a2;
  s2[tid] = a0 * a0 + a1 * a1 + a2 * a2;
  __syncthreads();
#pragma unroll
  for (int s = 128; s > 0; s >>= 1) {
    if (tid < s) { s1[tid] += s1[tid + s]; s2[tid] += s2[tid + s]; }
    __syncthreads();
  }
  const float mean = s1[0] * (1.0f / ND);
  const float var  = s2[0] * (1.0f / ND) - mean * mean;
  const float rstd = rsqrtf(var + LN_EPS);
  bf16* yr = xn + (size_t)m * ND;
  yr[tid]       = (bf16)((a0 - mean) * rstd * g[tid]       + b[tid]);
  yr[tid + 256] = (bf16)((a1 - mean) * rstd * g[tid + 256] + b[tid + 256]);
  yr[tid + 512] = (bf16)((a2 - mean) * rstd * g[tid + 512] + b[tid + 512]);
}

// ---- Kernel 2: transpose + convert weights to bf16 (Wt[c*R+r] = W[r*C+c]) -
__global__ void k_wt(const float* __restrict__ w, bf16* __restrict__ wt,
                     int R, int C) {
  int idx = blockIdx.x * 256 + threadIdx.x;
  if (idx < R * C) {
    int c = idx / R, r = idx % R;
    wt[idx] = (bf16)w[(size_t)r * C + c];
  }
}

// ---- Kernel 3: QKV GEMM (128x64 tile, K=768), double-buffered, ------------
//      per-head LayerNorm epilogue entirely in registers (shfl_xor).
// grid = (36 col-tiles, 128 row-tiles), block = 256 (8 waves).
__global__ void k_qkv(const bf16* __restrict__ xn, const bf16* __restrict__ wT,
                      const float* __restrict__ bqkv,
                      const float* __restrict__ qg, const float* __restrict__ qb,
                      const float* __restrict__ kg, const float* __restrict__ kb,
                      bf16* __restrict__ qws, bf16* __restrict__ kws,
                      bf16* __restrict__ vT) {
  __shared__ __align__(16) bf16 As2[2][128 * 40];
  __shared__ __align__(16) bf16 Bs2[2][64 * 40];

  const int ct = blockIdx.x;
  const int which = ct / NH, head = ct % NH;
  const int n0 = ct * 64;
  const int m0 = blockIdx.y * 128;
  const int tid = threadIdx.x, lane = tid & 31, wave = tid >> 5;
  const int ln = lane & 15, half = lane >> 4;
  const int ar = tid >> 1, apart = tid & 1;   // A-tile load assignment

  v8f acc[4];
#pragma unroll
  for (int j = 0; j < 4; ++j) { v8f z = {}; acc[j] = z; }

  // prologue: fill buffer 0
  {
    const bf16* ga = xn + (size_t)(m0 + ar) * ND + apart * 16;
    async_ld_b128(&As2[0][ar * 40 + apart * 16], ga);
    async_ld_b128(&As2[0][ar * 40 + apart * 16 + 8], ga + 8);
  }
  if (tid < 32)
    tdm_load_2d(lds_off(Bs2[0]), wT + (size_t)n0 * ND,
                32, 64, ND, 3 * ND, ND, 3, 3);
  wait_async0();
  if (tid < 32) wait_tensor0();
  __syncthreads();

  const int KT = ND / 32;  // 24
#pragma unroll 1           // hand-pipelined: keep one body, accs stay in place
  for (int kt = 0; kt < KT; ++kt) {
    const int cur = kt & 1, nxt = cur ^ 1;
    if (kt + 1 < KT) {  // prefetch next K-slice into the other buffer
      const int k1 = (kt + 1) * 32;
      const bf16* ga = xn + (size_t)(m0 + ar) * ND + k1 + apart * 16;
      async_ld_b128(&As2[nxt][ar * 40 + apart * 16], ga);
      async_ld_b128(&As2[nxt][ar * 40 + apart * 16 + 8], ga + 8);
      if (tid < 32)
        tdm_load_2d(lds_off(Bs2[nxt]), wT + (size_t)n0 * ND + k1,
                    32, 64, ND, 3 * ND, ND, 3, 3);
    }
    v16bf a = load_frag(As2[cur] + wave * 16 * 40, 40, lane);
#pragma unroll
    for (int j = 0; j < 4; ++j) {
      v16bf b = load_frag(Bs2[cur] + j * 16 * 40, 40, lane);
      acc[j] = wmma_bf16(a, b, acc[j]);
    }
    wait_async0();
    if (tid < 32) wait_tensor0();
    __syncthreads();
  }

  // Register epilogue: each wave's 4 accs hold all 64 head dims of its 16
  // rows; each row lives in one 16-lane half -> shfl_xor row reductions.
  float bias[4], gsel[4], bsel[4];
#pragma unroll
  for (int j = 0; j < 4; ++j) bias[j] = bqkv[n0 + j * 16 + ln];
  if (which != 2) {
    const float* gp = (which == 0) ? qg : kg;
    const float* bp = (which == 0) ? qb : kb;
#pragma unroll
    for (int j = 0; j < 4; ++j) { gsel[j] = gp[j * 16 + ln]; bsel[j] = bp[j * 16 + ln]; }
  }
#pragma unroll
  for (int v = 0; v < 8; ++v) {
    float xr[4];
#pragma unroll
    for (int j = 0; j < 4; ++j) xr[j] = acc[j][v] + bias[j];
    const int row = wave * 16 + v + 8 * half;
    const int m = m0 + row;
    const int bidx = m >> 10, nidx = m & 1023;
    const int bh = bidx * NH + head;
    if (which == 2) {  // V: store transposed [B,H,Hd,N] for the PV GEMM
#pragma unroll
      for (int j = 0; j < 4; ++j)
        vT[((size_t)bh * HD + j * 16 + ln) * NN + nidx] = (bf16)xr[j];
    } else {           // Q/K: per-head LayerNorm over Hd=64, then store
      float s = (xr[0] + xr[1]) + (xr[2] + xr[3]);
      float ss = xr[0] * xr[0] + xr[1] * xr[1] + xr[2] * xr[2] + xr[3] * xr[3];
#pragma unroll
      for (int msk = 1; msk < 16; msk <<= 1) {
        s  += __shfl_xor(s, msk, 32);
        ss += __shfl_xor(ss, msk, 32);
      }
      float mean = s * (1.0f / HD);
      float var  = ss * (1.0f / HD) - mean * mean;
      float rstd = rsqrtf(var + LN_EPS);
      bf16* dst = ((which == 0) ? qws : kws) + ((size_t)bh * NN + nidx) * HD;
#pragma unroll
      for (int j = 0; j < 4; ++j)
        dst[j * 16 + ln] = (bf16)((xr[j] - mean) * rstd * gsel[j] + bsel[j]);
    }
  }
}

// ---- Kernel 4: flash attention, TDM double-buffered K/V -------------------
// grid = (16 q-blocks of 64 rows, 192 (b,h)), block = 128 (4 waves x 16 rows).
__global__ void k_attn(const bf16* __restrict__ qws, const bf16* __restrict__ kws,
                       const bf16* __restrict__ vT, bf16* __restrict__ ctx) {
  __shared__ __align__(16) bf16 Qs[64 * 72];
  __shared__ __align__(16) bf16 Ks2[2][64 * 72];
  __shared__ __align__(16) bf16 Vts2[2][64 * 72];
  __shared__ __align__(16) bf16 Ps[4][16 * 72];

  const int qblk = blockIdx.x, bh = blockIdx.y;
  const int tid = threadIdx.x, lane = tid & 31, wave = tid >> 5;
  const int ln = lane & 15, half = lane >> 4;
  bf16* Psw = Ps[wave];

  const bf16* qbase = qws + ((size_t)bh * NN + qblk * 64) * HD;
  const bf16* kbase = kws + (size_t)bh * NN * HD;
  const bf16* vbase = vT + (size_t)bh * HD * NN;

  if (tid < 32) {  // rows are 128B; +16B pad -> pitch 144B (stride 72)
    tdm_load_2d(lds_off(Qs), qbase, HD, 64, HD, NN, HD, 4, 3);
    tdm_load_2d(lds_off(Ks2[0]), kbase, HD, KB, HD, NN, HD, 4, 3);
    tdm_load_2d(lds_off(Vts2[0]), vbase, KB, HD, NN, HD, NN, 4, 3);
    wait_tensor0();
  }
  float m_run[8], l_run[8];
  v8f accO[4];
#pragma unroll
  for (int v = 0; v < 8; ++v) { m_run[v] = -1e30f; l_run[v] = 0.f; }
#pragma unroll
  for (int j = 0; j < 4; ++j) { v8f z = {}; accO[j] = z; }
  __syncthreads();

  // Q fragments are loop-invariant: hoist
  v16bf aq0 = load_frag(Qs + wave * 16 * 72, 72, lane);
  v16bf aq1 = load_frag(Qs + wave * 16 * 72 + 32, 72, lane);

#pragma unroll 1           // hand-pipelined double buffer
  for (int jb = 0; jb < NBLK; ++jb) {
    const int cur = jb & 1, nxt = cur ^ 1;
    if (jb + 1 < NBLK && tid < 32) {  // prefetch next K/V block via TDM
      tdm_load_2d(lds_off(Ks2[nxt]), kbase + (size_t)(jb + 1) * KB * HD,
                  HD, KB, HD, NN, HD, 4, 3);
      tdm_load_2d(lds_off(Vts2[nxt]), vbase + (jb + 1) * KB,
                  KB, HD, NN, HD, NN, 4, 3);
    }
    const bf16* Kc = Ks2[cur];
    const bf16* Vc = Vts2[cur];

    // S = Q K^T  (16 q-rows x 64 keys per wave; Hd=64 -> 2 K-steps)
    v8f sacc[4];
#pragma unroll
    for (int s = 0; s < 4; ++s) {
      v16bf bk0 = load_frag(Kc + s * 16 * 72, 72, lane);
      v16bf bk1 = load_frag(Kc + s * 16 * 72 + 32, 72, lane);
      v8f z = {};
      sacc[s] = wmma_bf16(aq0, bk0, z);
      sacc[s] = wmma_bf16(aq1, bk1, sacc[s]);
    }

    // Online softmax (row reductions within the 16-lane half-groups)
#pragma unroll
    for (int v = 0; v < 8; ++v) {
      float x0 = sacc[0][v] * SCALE, x1 = sacc[1][v] * SCALE;
      float x2 = sacc[2][v] * SCALE, x3 = sacc[3][v] * SCALE;
      float mx = fmaxf(fmaxf(x0, x1), fmaxf(x2, x3));
#pragma unroll
      for (int msk = 1; msk < 16; msk <<= 1) mx = fmaxf(mx, __shfl_xor(mx, msk, 32));
      float mnew = fmaxf(m_run[v], mx);
      float corr = __expf(m_run[v] - mnew);
      float p0 = __expf(x0 - mnew), p1 = __expf(x1 - mnew);
      float p2 = __expf(x2 - mnew), p3 = __expf(x3 - mnew);
      float rs = (p0 + p1) + (p2 + p3);
#pragma unroll
      for (int msk = 1; msk < 16; msk <<= 1) rs += __shfl_xor(rs, msk, 32);
      l_run[v] = l_run[v] * corr + rs;
      m_run[v] = mnew;
#pragma unroll
      for (int j = 0; j < 4; ++j) accO[j][v] *= corr;
      int prow = v + 8 * half;
      Psw[prow * 72 + ln]      = (bf16)p0;   // LDS in-order within wave:
      Psw[prow * 72 + 16 + ln] = (bf16)p1;   // no barrier needed before the
      Psw[prow * 72 + 32 + ln] = (bf16)p2;   // P-fragment reload below
      Psw[prow * 72 + 48 + ln] = (bf16)p3;
    }

    // O += P V  (K=64 keys -> 2 K-steps, N=64 dims)
    v16bf pf0 = load_frag(Psw, 72, lane);
    v16bf pf1 = load_frag(Psw + 32, 72, lane);
#pragma unroll
    for (int j = 0; j < 4; ++j) {
      v16bf vf0 = load_frag(Vc + j * 16 * 72, 72, lane);
      v16bf vf1 = load_frag(Vc + j * 16 * 72 + 32, 72, lane);
      accO[j] = wmma_bf16(pf0, vf0, accO[j]);
      accO[j] = wmma_bf16(pf1, vf1, accO[j]);
    }

    if (jb + 1 < NBLK && tid < 32) wait_tensor0();  // next buffer landed
    __syncthreads();
  }

  // Normalize and scatter to ctx [B*N, 768] (token-major for proj GEMM)
  const int bidx = bh / NH, head = bh % NH;
#pragma unroll
  for (int v = 0; v < 8; ++v) {
    float inv = 1.0f / l_run[v];
    int n = qblk * 64 + wave * 16 + v + 8 * half;
    bf16* crow = ctx + ((size_t)bidx * NN + n) * ND + head * HD;
#pragma unroll
    for (int j = 0; j < 4; ++j)
      crow[j * 16 + ln] = (bf16)(accO[j][v] * inv);
  }
}

// ---- Kernel 5: proj GEMM (128x128 tile, K=768), double-buffered, f32 out --
__global__ void k_proj(const bf16* __restrict__ ctx, const bf16* __restrict__ wT,
                       const float* __restrict__ bias, float* __restrict__ out) {
  __shared__ __align__(16) bf16 As2[2][128 * 40];
  __shared__ __align__(16) bf16 Bs2[2][128 * 40];

  const int n0 = blockIdx.x * 128;
  const int m0 = blockIdx.y * 128;
  const int tid = threadIdx.x, lane = tid & 31, wave = tid >> 5;
  const int ln = lane & 15, half = lane >> 4;
  const int r = tid >> 1, part = tid & 1;

  v8f acc[8];
#pragma unroll
  for (int j = 0; j < 8; ++j) { v8f z = {}; acc[j] = z; }

  {  // prologue: fill buffer 0
    const bf16* ga = ctx + (size_t)(m0 + r) * ND + part * 16;
    async_ld_b128(&As2[0][r * 40 + part * 16], ga);
    async_ld_b128(&As2[0][r * 40 + part * 16 + 8], ga + 8);
    const bf16* gb = wT + (size_t)(n0 + r) * ND + part * 16;
    async_ld_b128(&Bs2[0][r * 40 + part * 16], gb);
    async_ld_b128(&Bs2[0][r * 40 + part * 16 + 8], gb + 8);
  }
  wait_async0();
  __syncthreads();

  const int KT = ND / 32;
#pragma unroll 1           // hand-pipelined double buffer
  for (int kt = 0; kt < KT; ++kt) {
    const int cur = kt & 1, nxt = cur ^ 1;
    if (kt + 1 < KT) {
      const int k1 = (kt + 1) * 32;
      const bf16* ga = ctx + (size_t)(m0 + r) * ND + k1 + part * 16;
      async_ld_b128(&As2[nxt][r * 40 + part * 16], ga);
      async_ld_b128(&As2[nxt][r * 40 + part * 16 + 8], ga + 8);
      const bf16* gb = wT + (size_t)(n0 + r) * ND + k1 + part * 16;
      async_ld_b128(&Bs2[nxt][r * 40 + part * 16], gb);
      async_ld_b128(&Bs2[nxt][r * 40 + part * 16 + 8], gb + 8);
    }
    v16bf a = load_frag(As2[cur] + wave * 16 * 40, 40, lane);
#pragma unroll
    for (int j = 0; j < 8; ++j) {
      v16bf b = load_frag(Bs2[cur] + j * 16 * 40, 40, lane);
      acc[j] = wmma_bf16(a, b, acc[j]);
    }
    wait_async0();
    __syncthreads();
  }

#pragma unroll
  for (int j = 0; j < 8; ++j) {
    float bv = bias[n0 + j * 16 + ln];
#pragma unroll
    for (int e = 0; e < 8; ++e) {
      int row = m0 + wave * 16 + e + 8 * half;
      out[(size_t)row * ND + n0 + j * 16 + ln] = acc[j][e] + bv;
    }
  }
}

// ---------------------------------------------------------------------------
extern "C" void kernel_launch(void* const* d_in, const int* in_sizes, int n_in,
                              void* d_out, int out_size, void* d_ws, size_t ws_size,
                              hipStream_t stream) {
  const float* x      = (const float*)d_in[0];
  const float* ln_g   = (const float*)d_in[1];
  const float* ln_b   = (const float*)d_in[2];
  const float* W_qkv  = (const float*)d_in[3];
  const float* b_qkv  = (const float*)d_in[4];
  const float* qn_g   = (const float*)d_in[5];
  const float* qn_b   = (const float*)d_in[6];
  const float* kn_g   = (const float*)d_in[7];
  const float* kn_b   = (const float*)d_in[8];
  const float* W_proj = (const float*)d_in[9];
  const float* b_proj = (const float*)d_in[10];
  float* out = (float*)d_out;

  char* ws = (char*)d_ws;
  size_t off = 0;
  bf16* xn     = (bf16*)(ws + off); off += (size_t)BN * ND * 2;
  bf16* wqkvT  = (bf16*)(ws + off); off += (size_t)3 * ND * ND * 2;
  bf16* wprojT = (bf16*)(ws + off); off += (size_t)ND * ND * 2;
  bf16* qws    = (bf16*)(ws + off); off += (size_t)BN * ND * 2;
  bf16* kws    = (bf16*)(ws + off); off += (size_t)BN * ND * 2;
  bf16* vT     = (bf16*)(ws + off); off += (size_t)BN * ND * 2;
  bf16* ctx    = (bf16*)(ws + off); off += (size_t)BN * ND * 2;

  k_ln_x<<<BN, 256, 0, stream>>>(x, ln_g, ln_b, xn);
  k_wt<<<(ND * 3 * ND + 255) / 256, 256, 0, stream>>>(W_qkv, wqkvT, ND, 3 * ND);
  k_wt<<<(ND * ND + 255) / 256, 256, 0, stream>>>(W_proj, wprojT, ND, ND);
  k_qkv<<<dim3(36, BN / 128), 256, 0, stream>>>(xn, wqkvT, b_qkv,
                                                qn_g, qn_b, kn_g, kn_b,
                                                qws, kws, vT);
  k_attn<<<dim3(NN / 64, NB * NH), 128, 0, stream>>>(qws, kws, vT, ctx);
  k_proj<<<dim3(ND / 128, BN / 128), 256, 0, stream>>>(ctx, wprojT, b_proj, out);
}
